// NanoDeepSeek_1331439862378
// MI455X (gfx1250) — matrix operand order repaired and checked by reading, not compiled, not run on hardware
//
#include <hip/hip_runtime.h>
#include <math.h>

// ---------------------------------------------------------------------------
// NanoDeepSeek forward for MI455X (gfx1250, wave32, WMMA).
// bf16 WMMA (v_wmma_f32_16x16x32_bf16) with f32 accumulate; fp32 operands are
// converted to bf16 while staging tiles into LDS. Tile fetch is software-
// pipelined through registers so global-load latency overlaps the WMMAs.
// Fused epilogues: alpha, exact-erf GELU, per-row scale (MoE combine),
// residual add, accumulate.
// ---------------------------------------------------------------------------

typedef __attribute__((ext_vector_type(16))) __bf16 v16bf;
typedef __attribute__((ext_vector_type(8)))  __bf16 bf16x8;
typedef __attribute__((ext_vector_type(4)))  __bf16 bf16x4;
typedef __attribute__((ext_vector_type(8)))  float  v8f;

// Model dims
#define SEQ     2048
#define HDIM    1024
#define CDIM    256
#define NHEADS  16
#define DHEAD   64
#define DROPE   32
#define UPDIM   512
#define EDIM    4096
#define NEXP    8
#define NVOCAB  32000

enum { GF_TRANSB = 1, GF_GELU = 2, GF_ACCUM = 4, GF_ADDRES = 8, GF_ROWSCALE = 16 };

// ---------------------------------------------------------------------------
// Generic GEMM: C[2048 x N] = epilogue( alpha * A[2048 x K] @ op(B) )
// Block: 256 threads (8 waves). Block tile 128x64, K-step 32.
// Wave tile 32x32 = 2x2 WMMA 16x16 subtiles.
// LDS stride 40 bf16 (80B) -> conflict-free b128 fragment reads.
// ---------------------------------------------------------------------------
__launch_bounds__(256)
__global__ void gemm_bf16_wmma(const float* __restrict__ A, int lda,
                               const float* __restrict__ B, int ldb,
                               float* __restrict__ C, int ldc,
                               int K, float alpha, int flags,
                               const float* __restrict__ rowscale, int rsStride,
                               const float* __restrict__ res, int ldres)
{
    __shared__ __bf16 ldsA[128 * 40];   // [m][k]
    __shared__ __bf16 ldsB[64 * 40];    // transposed: [n][k]

    const int tid  = threadIdx.x;
    const int lane = tid & 31;
    const int wave = tid >> 5;
    const int wm   = wave & 3;          // wave M position (0..3)
    const int wn   = wave >> 2;         // wave N position (0..1)
    const int half = lane >> 4;         // lane group (0 or 1)
    const int lrow = lane & 15;
    const int m0 = blockIdx.y * 128;
    const int n0 = blockIdx.x * 64;

    v8f acc[2][2] = {};

    // register staging buffers for the software pipeline
    float4 ra[4], rb[2];

    auto load_tile = [&](int k0) {
        #pragma unroll
        for (int j = 0; j < 4; ++j) {
            const int fid = tid + j * 256;          // 1024 float4 total
            const int r   = fid >> 3;
            const int c4  = (fid & 7) << 2;
            ra[j] = *(const float4*)(A + (size_t)(m0 + r) * lda + k0 + c4);
        }
        if (flags & GF_TRANSB) {
            #pragma unroll
            for (int j = 0; j < 2; ++j) {
                const int fid = tid + j * 256;      // 512 float4 total
                const int r   = fid >> 3;           // n (0..63)
                const int c4  = (fid & 7) << 2;     // k
                rb[j] = *(const float4*)(B + (size_t)(n0 + r) * ldb + k0 + c4);
            }
        } else {
            #pragma unroll
            for (int j = 0; j < 2; ++j) {
                const int fid = tid + j * 256;
                const int r   = fid >> 4;           // k (0..31)
                const int c4  = (fid & 15) << 2;    // n
                rb[j] = *(const float4*)(B + (size_t)(k0 + r) * ldb + n0 + c4);
            }
        }
    };

    auto store_tile = [&]() {
        #pragma unroll
        for (int j = 0; j < 4; ++j) {
            const int fid = tid + j * 256;
            const int r   = fid >> 3;
            const int c4  = (fid & 7) << 2;
            bf16x4 b;
            b.x = (__bf16)ra[j].x; b.y = (__bf16)ra[j].y;
            b.z = (__bf16)ra[j].z; b.w = (__bf16)ra[j].w;
            *(bf16x4*)&ldsA[r * 40 + c4] = b;
        }
        if (flags & GF_TRANSB) {
            #pragma unroll
            for (int j = 0; j < 2; ++j) {
                const int fid = tid + j * 256;
                const int r   = fid >> 3;
                const int c4  = (fid & 7) << 2;
                bf16x4 b;
                b.x = (__bf16)rb[j].x; b.y = (__bf16)rb[j].y;
                b.z = (__bf16)rb[j].z; b.w = (__bf16)rb[j].w;
                *(bf16x4*)&ldsB[r * 40 + c4] = b;
            }
        } else {
            #pragma unroll
            for (int j = 0; j < 2; ++j) {
                const int fid = tid + j * 256;
                const int r   = fid >> 4;
                const int c4  = (fid & 15) << 2;
                ldsB[(c4 + 0) * 40 + r] = (__bf16)rb[j].x;
                ldsB[(c4 + 1) * 40 + r] = (__bf16)rb[j].y;
                ldsB[(c4 + 2) * 40 + r] = (__bf16)rb[j].z;
                ldsB[(c4 + 3) * 40 + r] = (__bf16)rb[j].w;
            }
        }
    };

    // ---- prologue: fetch tile 0 into registers ----
    load_tile(0);

    for (int k0 = 0; k0 < K; k0 += 32) {
        // commit current tile (registers -> LDS, fp32 -> bf16)
        store_tile();
        __syncthreads();

        // issue next tile's global loads; latency hides behind the WMMAs
        if (k0 + 32 < K) load_tile(k0 + 32);

        // warm L2 two tiles ahead (global_prefetch_b8)
        if (k0 + 64 < K) {
            __builtin_prefetch(A + (size_t)(m0 + (tid >> 1)) * lda + k0 + 64, 0, 1);
            const float* bp = (flags & GF_TRANSB)
                ? B + (size_t)(n0 + (tid & 63)) * ldb + k0 + 64
                : B + (size_t)(k0 + 64 + (tid & 31)) * ldb + n0;
            __builtin_prefetch(bp, 0, 1);
        }

        // -------- fragments per ISA 16-bit layouts --------
        // A 16x32: lane<16 -> K{0..7},{16..23}; lane>=16 -> K{8..15},{24..31}
        v16bf af[2], bfr[2];
        #pragma unroll
        for (int i = 0; i < 2; ++i) {
            const int rba = (wm * 32 + i * 16 + lrow) * 40;
            bf16x8 lo = *(const bf16x8*)&ldsA[rba + half * 8];
            bf16x8 hi = *(const bf16x8*)&ldsA[rba + 16 + half * 8];
            af[i] = __builtin_shufflevector(lo, hi, 0,1,2,3,4,5,6,7,8,9,10,11,12,13,14,15);
        }
        // B 32x16: lane<16 -> N=lane, K{0..7},{16..23}; lane>=16 -> N=lane-16, K{8..15},{24..31}
        #pragma unroll
        for (int j = 0; j < 2; ++j) {
            const int rbb = (wn * 32 + j * 16 + lrow) * 40 + half * 8;
            bf16x8 lo = *(const bf16x8*)&ldsB[rbb];
            bf16x8 hi = *(const bf16x8*)&ldsB[rbb + 16];
            bfr[j] = __builtin_shufflevector(lo, hi, 0,1,2,3,4,5,6,7,8,9,10,11,12,13,14,15);
        }

        #pragma unroll
        for (int i = 0; i < 2; ++i)
            #pragma unroll
            for (int j = 0; j < 2; ++j)
                acc[i][j] = __builtin_amdgcn_wmma_f32_16x16x32_bf16(
                    false, af[i], false, bfr[j], (short)0, acc[i][j], false, false);

        // all waves done reading LDS before the next store_tile overwrites it
        __syncthreads();
    }

    // -------- epilogue (C/D layout: lane<16 M=r, lane>=16 M=8+r) --------
    #pragma unroll
    for (int i = 0; i < 2; ++i) {
        #pragma unroll
        for (int j = 0; j < 2; ++j) {
            const int col   = n0 + wn * 32 + j * 16 + lrow;
            const int rbase = m0 + wm * 32 + i * 16 + half * 8;
            #pragma unroll
            for (int r = 0; r < 8; ++r) {
                const int row = rbase + r;
                float v = acc[i][j][r] * alpha;
                if (flags & GF_GELU)
                    v = 0.5f * v * (1.0f + erff(v * 0.7071067811865475f));
                if (flags & GF_ROWSCALE)
                    v *= rowscale[(size_t)row * rsStride];
                if (flags & GF_ADDRES)
                    v += res[(size_t)row * ldres + col];
                float* cp = C + (size_t)row * ldc + col;
                if (flags & GF_ACCUM) *cp += v; else *cp = v;
            }
        }
    }
}

// ---------------------------------------------------------------------------
// Embedding gather: out[s][:] = emb[tokens[s]][:]
// ---------------------------------------------------------------------------
__launch_bounds__(256)
__global__ void embed_k(const int* __restrict__ tokens,
                        const float* __restrict__ emb,
                        float* __restrict__ out)
{
    const int s = blockIdx.x;
    const int tok = tokens[s];
    ((float4*)out)[(size_t)s * 256 + threadIdx.x] =
        ((const float4*)emb)[(size_t)tok * 256 + threadIdx.x];
}

// ---------------------------------------------------------------------------
// LayerNorm over 1024 features, one wave per row.
// ---------------------------------------------------------------------------
__launch_bounds__(256)
__global__ void layernorm_k(const float* __restrict__ x,
                            const float* __restrict__ scale,
                            const float* __restrict__ bias,
                            float* __restrict__ out)
{
    const int row  = blockIdx.x * 8 + (threadIdx.x >> 5);
    const int lane = threadIdx.x & 31;
    const float4* xr = (const float4*)(x + (size_t)row * HDIM);
    float4 v[8];
    float sum = 0.f, sq = 0.f;
    #pragma unroll
    for (int j = 0; j < 8; ++j) {
        v[j] = xr[lane + 32 * j];
        sum += v[j].x + v[j].y + v[j].z + v[j].w;
        sq  += v[j].x * v[j].x + v[j].y * v[j].y + v[j].z * v[j].z + v[j].w * v[j].w;
    }
    #pragma unroll
    for (int off = 16; off; off >>= 1) {
        sum += __shfl_xor(sum, off, 32);
        sq  += __shfl_xor(sq,  off, 32);
    }
    const float mean = sum * (1.0f / HDIM);
    const float var  = sq * (1.0f / HDIM) - mean * mean;
    const float rstd = rsqrtf(var + 1e-5f);
    float4* orow = (float4*)(out + (size_t)row * HDIM);
    const float4* s4 = (const float4*)scale;
    const float4* b4 = (const float4*)bias;
    #pragma unroll
    for (int j = 0; j < 8; ++j) {
        const float4 sv = s4[lane + 32 * j];
        const float4 bv = b4[lane + 32 * j];
        float4 o;
        o.x = (v[j].x - mean) * rstd * sv.x + bv.x;
        o.y = (v[j].y - mean) * rstd * sv.y + bv.y;
        o.z = (v[j].z - mean) * rstd * sv.z + bv.z;
        o.w = (v[j].w - mean) * rstd * sv.w + bv.w;
        orow[lane + 32 * j] = o;
    }
}

// ---------------------------------------------------------------------------
// RoPE + assembly: out[s,h,0:32]=cpart, out[s,h,32:48]=rot(rpart[0:16]),
// out[s,h,48:64]=rpart[16:32]. pos = s/40, inv_freq[j]=10000^{-j/8}.
// ---------------------------------------------------------------------------
__launch_bounds__(256)
__global__ void rope_assemble_k(const float* __restrict__ cpart,
                                const float* __restrict__ rpart,
                                float* __restrict__ out)
{
    const int s = blockIdx.x;
    const float pos = (float)s / 40.0f;
    for (int t = threadIdx.x; t < 1024; t += 256) {
        const int h = t >> 6, d = t & 63;
        float val;
        if (d < 32) {
            val = cpart[(size_t)s * 512 + h * 32 + d];
        } else {
            const int i = d - 32;
            if (i < 16) {
                const int jj = i & 7;
                const float f = pos * powf(10000.0f, -(float)jj * 0.125f);
                const float xr = rpart[(size_t)s * 512 + h * 32 + i];
                const float xo = (i < 8) ? -rpart[(size_t)s * 512 + h * 32 + i + 8]
                                         :  rpart[(size_t)s * 512 + h * 32 + i - 8];
                val = xr * cosf(f) + xo * sinf(f);
            } else {
                val = rpart[(size_t)s * 512 + h * 32 + i];
            }
        }
        out[(size_t)s * 1024 + t] = val;
    }
}

// ---------------------------------------------------------------------------
// Router: logits = x2 @ Wr (1024x8), softmax, top-3 -> dense comb[2048][8].
// One wave per row.
// ---------------------------------------------------------------------------
__launch_bounds__(256)
__global__ void router_k(const float* __restrict__ x2,
                         const float* __restrict__ Wr,
                         float* __restrict__ comb)
{
    const int row  = blockIdx.x * 8 + (threadIdx.x >> 5);
    const int lane = threadIdx.x & 31;
    float acc[8] = {};
    for (int k = lane; k < HDIM; k += 32) {
        const float xv = x2[(size_t)row * HDIM + k];
        const float4* wr = (const float4*)(Wr + (size_t)k * 8);
        const float4 a = wr[0], b = wr[1];
        acc[0] += xv * a.x; acc[1] += xv * a.y; acc[2] += xv * a.z; acc[3] += xv * a.w;
        acc[4] += xv * b.x; acc[5] += xv * b.y; acc[6] += xv * b.z; acc[7] += xv * b.w;
    }
    #pragma unroll
    for (int e = 0; e < 8; ++e)
        #pragma unroll
        for (int off = 16; off; off >>= 1)
            acc[e] += __shfl_xor(acc[e], off, 32);
    if (lane == 0) {
        float m = acc[0];
        #pragma unroll
        for (int e = 1; e < 8; ++e) m = fmaxf(m, acc[e]);
        float p[8], sum = 0.f;
        #pragma unroll
        for (int e = 0; e < 8; ++e) { p[e] = expf(acc[e] - m); sum += p[e]; }
        const float inv = 1.0f / sum;
        #pragma unroll
        for (int e = 0; e < 8; ++e) p[e] *= inv;
        float cb[8] = {};
        for (int t = 0; t < 3; ++t) {
            int bi = 0; float bv = -1.0f;
            #pragma unroll
            for (int e = 0; e < 8; ++e) if (p[e] > bv) { bv = p[e]; bi = e; }
            cb[bi] = p[bi];
            p[bi] = -2.0f;
        }
        #pragma unroll
        for (int e = 0; e < 8; ++e) comb[(size_t)row * 8 + e] = cb[e];
    }
}

// ---------------------------------------------------------------------------
// Elementwise helpers (float4 granularity)
// ---------------------------------------------------------------------------
__launch_bounds__(256)
__global__ void copy_f4(const float4* __restrict__ a, float4* __restrict__ o)
{
    const int i = blockIdx.x * 256 + threadIdx.x;
    o[i] = a[i];
}

__launch_bounds__(256)
__global__ void add_f4(float4* __restrict__ o, const float4* __restrict__ a)
{
    const int i = blockIdx.x * 256 + threadIdx.x;
    float4 v = o[i];
    const float4 b = a[i];
    v.x += b.x; v.y += b.y; v.z += b.z; v.w += b.w;
    o[i] = v;
}

// ---------------------------------------------------------------------------
// Host orchestration
// ---------------------------------------------------------------------------
extern "C" void kernel_launch(void* const* d_in, const int* in_sizes, int n_in,
                              void* d_out, int out_size, void* d_ws, size_t ws_size,
                              hipStream_t stream)
{
    (void)in_sizes; (void)n_in; (void)out_size; (void)ws_size;

    const int*   tokens   = (const int*)  d_in[0];
    const float* ln1_s    = (const float*)d_in[1];
    const float* ln1_b    = (const float*)d_in[2];
    const float* W_dkv    = (const float*)d_in[3];
    const float* W_dq     = (const float*)d_in[4];
    const float* W_uk     = (const float*)d_in[5];
    const float* W_uv     = (const float*)d_in[6];
    const float* W_uq     = (const float*)d_in[7];
    const float* W_kr     = (const float*)d_in[8];
    const float* W_qr     = (const float*)d_in[9];
    const float* W_out    = (const float*)d_in[10];
    const float* ln2_s    = (const float*)d_in[11];
    const float* ln2_b    = (const float*)d_in[12];
    const float* W_su     = (const float*)d_in[13];
    const float* W_sd     = (const float*)d_in[14];
    const float* W_eu     = (const float*)d_in[15];
    const float* W_ed     = (const float*)d_in[16];
    const float* W_router = (const float*)d_in[17];
    const float* emb      = (const float*)d_in[18];
    const float* lnf_s    = (const float*)d_in[19];
    const float* lnf_b    = (const float*)d_in[20];
    const float* W_head   = (const float*)d_in[21];

    float* ws = (float*)d_ws;
    const size_t M1 = 1u << 20;               // 1M floats
    float* H    = ws + 0;                     // 2M  running hidden
    float* LN   = ws + 2 * M1;                // 2M  ln output (x1 / x2 / final)
    float* CKV  = ws + 4 * M1;                // 0.5M
    float* CQ   = ws + 4 * M1 + (M1 >> 1);    // 0.5M
    float* QC   = ws + 5 * M1;                // 1M
    float* KC   = ws + 6 * M1;                // 1M
    float* Vb   = ws + 7 * M1;                // 2M
    float* KR   = ws + 9 * M1;                // 1M
    float* QR   = ws + 10 * M1;               // 1M
    float* Kb   = ws + 11 * M1;               // 2M  assembled K (s,h,d)
    float* Qb   = ws + 13 * M1;               // 2M  assembled Q
    float* SC   = ws + 15 * M1;               // 4M  per-head scores 2048x2048
    float* AO   = ws + 19 * M1;               // 2M  attn out (s,h,d)
    float* AT   = ws + 21 * M1;               // 2M  attn + residual
    float* UP   = ws + 23 * M1;               // 8M  expert up 2048x4096
    float* ACCB = ws + 31 * M1;               // 2M  moe accumulator
    float* COMB = ws + 33 * M1;               // 16K comb weights

    auto gemm = [&](const float* A, int lda, const float* B, int ldb,
                    float* C, int ldc, int N, int K, float alpha, int flags,
                    const float* rs, int rsStride, const float* res, int ldres) {
        dim3 g(N / 64, SEQ / 128);
        gemm_bf16_wmma<<<g, dim3(256), 0, stream>>>(A, lda, B, ldb, C, ldc,
                                                    K, alpha, flags, rs, rsStride, res, ldres);
    };

    // Embedding
    embed_k<<<SEQ, 256, 0, stream>>>(tokens, emb, H);

    for (int l = 0; l < 2; ++l) {
        const size_t lH  = (size_t)l * HDIM;

        // x1 = ln1(h)
        layernorm_k<<<SEQ / 8, 256, 0, stream>>>(H, ln1_s + lH, ln1_b + lH, LN);

        // low-rank projections
        gemm(LN, HDIM, W_dkv + (size_t)l * HDIM * CDIM, CDIM, CKV, CDIM, CDIM, HDIM, 1.f, 0, nullptr, 0, nullptr, 0);
        gemm(LN, HDIM, W_dq  + (size_t)l * HDIM * CDIM, CDIM, CQ,  CDIM, CDIM, HDIM, 1.f, 0, nullptr, 0, nullptr, 0);
        gemm(CQ,  CDIM, W_uq + (size_t)l * CDIM * UPDIM, UPDIM, QC, UPDIM, UPDIM, CDIM, 1.f, 0, nullptr, 0, nullptr, 0);
        gemm(CKV, CDIM, W_uk + (size_t)l * CDIM * UPDIM, UPDIM, KC, UPDIM, UPDIM, CDIM, 1.f, 0, nullptr, 0, nullptr, 0);
        gemm(CKV, CDIM, W_uv + (size_t)l * CDIM * HDIM,  HDIM,  Vb, HDIM,  HDIM,  CDIM, 1.f, 0, nullptr, 0, nullptr, 0);
        gemm(LN,  HDIM, W_kr + (size_t)l * HDIM * UPDIM, UPDIM, KR, UPDIM, UPDIM, HDIM, 1.f, 0, nullptr, 0, nullptr, 0);
        gemm(CQ,  CDIM, W_qr + (size_t)l * CDIM * UPDIM, UPDIM, QR, UPDIM, UPDIM, CDIM, 1.f, 0, nullptr, 0, nullptr, 0);

        // RoPE + concat into Q, K  (s, h, 64)
        rope_assemble_k<<<SEQ, 256, 0, stream>>>(QC, QR, Qb);
        rope_assemble_k<<<SEQ, 256, 0, stream>>>(KC, KR, Kb);

        // Attention per head: S = (K_h @ Q_h^T)/8 ; Out_h = S @ V_h   (no softmax, per reference)
        for (int h = 0; h < NHEADS; ++h) {
            gemm(Kb + h * DHEAD, HDIM, Qb + h * DHEAD, HDIM, SC, SEQ,
                 SEQ, DHEAD, 0.125f, GF_TRANSB, nullptr, 0, nullptr, 0);
            gemm(SC, SEQ, Vb + h * DHEAD, HDIM, AO + h * DHEAD, HDIM,
                 DHEAD, SEQ, 1.f, 0, nullptr, 0, nullptr, 0);
        }

        // attn = AO @ W_out + h
        gemm(AO, HDIM, W_out + (size_t)l * HDIM * HDIM, HDIM, AT, HDIM,
             HDIM, HDIM, 1.f, GF_ADDRES, nullptr, 0, H, HDIM);

        // x2 = ln2(attn); moe residual term: acc = x2
        layernorm_k<<<SEQ / 8, 256, 0, stream>>>(AT, ln2_s + lH, ln2_b + lH, LN);
        copy_f4<<<(SEQ * HDIM / 4) / 256, 256, 0, stream>>>((const float4*)LN, (float4*)ACCB);

        // router -> comb
        router_k<<<SEQ / 8, 256, 0, stream>>>(LN, W_router + (size_t)l * HDIM * NEXP, COMB);

        // shared expert: acc += gelu(x2 @ W_su) @ W_sd
        gemm(LN, HDIM, W_su + (size_t)l * HDIM * EDIM, EDIM, UP, EDIM,
             EDIM, HDIM, 1.f, GF_GELU, nullptr, 0, nullptr, 0);
        gemm(UP, EDIM, W_sd + (size_t)l * EDIM * HDIM, HDIM, ACCB, HDIM,
             HDIM, EDIM, 1.f, GF_ACCUM, nullptr, 0, nullptr, 0);

        // routed experts: acc += comb[:,e] * (gelu(x2 @ Weu) @ Wed)
        for (int e = 0; e < NEXP; ++e) {
            const size_t wo = ((size_t)l * NEXP + e) * (size_t)HDIM * EDIM;
            gemm(LN, HDIM, W_eu + wo, EDIM, UP, EDIM,
                 EDIM, HDIM, 1.f, GF_GELU, nullptr, 0, nullptr, 0);
            gemm(UP, EDIM, W_ed + wo, HDIM, ACCB, HDIM,
                 HDIM, EDIM, 1.f, GF_ACCUM | GF_ROWSCALE, COMB + e, NEXP, nullptr, 0);
        }

        // block output: h = h + (shared + routed + x2)
        add_f4<<<(SEQ * HDIM / 4) / 256, 256, 0, stream>>>((float4*)H, (const float4*)ACCB);
    }

    // final LN + LM head
    layernorm_k<<<SEQ / 8, 256, 0, stream>>>(H, lnf_s, lnf_b, LN);
    gemm(LN, HDIM, W_head, NVOCAB, (float*)d_out, NVOCAB,
         NVOCAB, HDIM, 1.f, 0, nullptr, 0, nullptr, 0);
}
